// GNNEncoder_64879775973499
// MI455X (gfx1250) — compile-verified, matrix operand order checked by reading, not run on hardware
//
#include <hip/hip_runtime.h>
#include <hip/hip_bf16.h>
#include <stdint.h>
#include <stddef.h>

typedef __attribute__((ext_vector_type(16))) __bf16 v16bf;
typedef __attribute__((ext_vector_type(8)))  float  v8f;
typedef __attribute__((ext_vector_type(4)))  int    v4i;

#define LN_EPS 1e-5f
#define AS1 __attribute__((address_space(1)))
#define AS3 __attribute__((address_space(3)))

#if __has_builtin(__builtin_amdgcn_global_load_async_to_lds_b128)
#define USE_ASYNC_LDS 1
#else
#define USE_ASYNC_LDS 0
#endif

// async global->LDS 16B copy (CDNA5 ASYNCcnt path); sync fallback
__device__ __forceinline__ void async_g2l_b128(const void* g, void* l) {
#if USE_ASYNC_LDS
  __builtin_amdgcn_global_load_async_to_lds_b128((AS1 v4i*)g, (AS3 v4i*)l, 0, 0);
#else
  *(uint4*)l = *(const uint4*)g;
#endif
}

__device__ __forceinline__ void wait_async0() {
#if __has_builtin(__builtin_amdgcn_s_wait_asynccnt)
  __builtin_amdgcn_s_wait_asynccnt(0);
#else
  asm volatile("s_wait_asynccnt 0x0" ::: "memory");
#endif
}

// ---------------------------------------------------------------- utilities
__global__ __launch_bounds__(256) void k_f32_to_bf16(const float* __restrict__ s,
                                                     __bf16* __restrict__ d, int n) {
  int i = blockIdx.x * 256 + threadIdx.x;
  if (i < n) d[i] = (__bf16)s[i];
}

__global__ __launch_bounds__(256) void k_fill1(float* __restrict__ p, int n) {
  int i = blockIdx.x * 256 + threadIdx.x;
  if (i < n) p[i] = 1.0f;
}

__global__ __launch_bounds__(256) void k_deg_scatter(const long long* __restrict__ dst,
                                                     float* __restrict__ deg, int n) {
  int i = blockIdx.x * 256 + threadIdx.x;
  if (i < n) atomicAdd(&deg[(int)dst[i]], 1.0f);
}

__global__ __launch_bounds__(256) void k_rsqrt_(float* __restrict__ p, int n) {
  int i = blockIdx.x * 256 + threadIdx.x;
  if (i < n) p[i] = rsqrtf(p[i]);
}

// edge MLP layer 1: t = relu(edge_attr @ We1 + be1)  (K=3, memory bound) -> bf16
__global__ __launch_bounds__(256) void k_edge_mlp1(const float* __restrict__ ea,
                                                   const float* __restrict__ W,
                                                   const float* __restrict__ b,
                                                   __bf16* __restrict__ t, int nN) {
  int idx = blockIdx.x * 256 + threadIdx.x;
  if (idx >= nN * 256) return;
  int i = idx >> 8, c = idx & 255;
  float e0 = ea[i * 3 + 0], e1 = ea[i * 3 + 1], e2 = ea[i * 3 + 2];
  float y = fmaf(e0, W[c], fmaf(e1, W[256 + c], fmaf(e2, W[512 + c], b[c])));
  t[idx] = (__bf16)fmaxf(y, 0.f);
}

// ---------------------------------------------------------------- WMMA GEMM
// C[M,KO] = A[M,K] @ W[K,KO](bf16) (+bias)(+relu), optional bf16 shadow output.
// ABF16=false: A fp32, converted to bf16 while staging through LDS.
// ABF16=true:  A bf16, tile DMA'd global->LDS via async copy (ASYNCcnt).
// block: 256 threads = 8 waves; tile 128(M) x 64(N); K step 32.
template <bool ABF16>
__global__ __launch_bounds__(256) void k_gemm(
    const void* __restrict__ Av, const __bf16* __restrict__ W,
    const float* __restrict__ bias, float* __restrict__ C,
    __bf16* __restrict__ Cbf, int M, int K, int KO, int relu) {
  __shared__ __bf16 As[128 * 40];   // A tile, padded stride (80B rows, 16B aligned)
  __shared__ __bf16 Bst[64 * 34];   // B tile transposed [N][K], padded stride

  const float*  Af = (const float*)Av;
  const __bf16* Ab = (const __bf16*)Av;

  const int tid  = threadIdx.x;
  const int lane = tid & 31;
  const int wm   = tid >> 5;                 // wave id -> M sub-tile
  const int bm0  = blockIdx.x * 128;
  const int n0   = blockIdx.y * 64;
  const int koff = (lane >= 16) ? 8 : 0;     // ISA half-wave K offset

  v8f cacc[4] = {};

  const int rt = tid >> 1;                   // staging row (0..127)
  const int cb = (tid & 1) * 16;             // staging col half (elements)
  const int ksteps = K / 32;

  for (int kt = 0; kt < ksteps; ++kt) {
    __syncthreads();
    // ---- stage A tile
    if (ABF16) {
      const size_t grow = (size_t)bm0 + rt;
      if (grow < (size_t)M) {
        const __bf16* gp = Ab + grow * (size_t)K + kt * 32 + cb;
        __bf16* lp = &As[rt * 40 + cb];
        async_g2l_b128(gp, lp);          // 8 bf16
        async_g2l_b128(gp + 8, lp + 8);  // 8 bf16
      }
    } else {
      float buf[16];
      const size_t grow = (size_t)bm0 + rt;
      if (grow < (size_t)M) {
        const float4* ap = (const float4*)(Af + grow * (size_t)K + kt * 32 + cb);
        if (kt + 1 < ksteps) __builtin_prefetch(ap + 8, 0, 0);  // next K tile
#pragma unroll
        for (int q = 0; q < 4; q++) {
          float4 v = ap[q];
          buf[4 * q + 0] = v.x; buf[4 * q + 1] = v.y;
          buf[4 * q + 2] = v.z; buf[4 * q + 3] = v.w;
        }
      } else {
#pragma unroll
        for (int q = 0; q < 16; q++) buf[q] = 0.f;
      }
#pragma unroll
      for (int q = 0; q < 16; q++) As[rt * 40 + cb + q] = (__bf16)buf[q];
    }
    // ---- stage W tile transposed (bf16 weights are L2-resident)
    {
      const int rk = tid >> 3;               // K row 0..31
      const int cn = (tid & 7) * 8;          // N col 0..56
      const uint4 w4 = *(const uint4*)(W + ((size_t)(kt * 32 + rk)) * KO + n0 + cn);
      const __bf16* pw = (const __bf16*)&w4;
#pragma unroll
      for (int q = 0; q < 8; q++) Bst[(cn + q) * 34 + rk] = pw[q];
    }
    if (ABF16) wait_async0();
    __syncthreads();

    // ---- build A fragment (16x32 bf16 per ISA layout), paired b32 LDS reads
    v16bf a;
    {
      const __bf16* ap = &As[(wm * 16 + (lane & 15)) * 40 + koff];
#pragma unroll
      for (int jj = 0; jj < 8; jj++) {
        int Kk = 2 * jj + ((jj >= 4) ? 8 : 0);
        unsigned int u = *(const unsigned int*)(ap + Kk);
        __bf16 h2[2];
        __builtin_memcpy(h2, &u, 4);
        a[2 * jj]     = h2[0];
        a[2 * jj + 1] = h2[1];
      }
    }
    // ---- build all 4 B fragments first, then issue WMMAs back-to-back
    v16bf bfr[4];
#pragma unroll
    for (int nf = 0; nf < 4; nf++) {
      const __bf16* bp_ = &Bst[(nf * 16 + (lane & 15)) * 34 + koff];
#pragma unroll
      for (int jj = 0; jj < 8; jj++) {
        int Kk = 2 * jj + ((jj >= 4) ? 8 : 0);
        unsigned int u = *(const unsigned int*)(bp_ + Kk);
        __bf16 h2[2];
        __builtin_memcpy(h2, &u, 4);
        bfr[nf][2 * jj]     = h2[0];
        bfr[nf][2 * jj + 1] = h2[1];
      }
    }
#pragma unroll
    for (int nf = 0; nf < 4; nf++)
      cacc[nf] = __builtin_amdgcn_wmma_f32_16x16x32_bf16(
          false, a, false, bfr[nf], (short)0, cacc[nf], false, false);
  }

  // ---- epilogue (C/D layout: VGPR r -> M=r(+8 upper half-wave), N=lane&15)
#pragma unroll
  for (int nf = 0; nf < 4; nf++) {
    const int col = n0 + nf * 16 + (lane & 15);
    const float bv = bias ? bias[col] : 0.f;
#pragma unroll
    for (int r = 0; r < 8; r++) {
      const int grow = bm0 + wm * 16 + r + ((lane >= 16) ? 8 : 0);
      if (grow < M) {
        float y = cacc[nf][r] + bv;
        if (relu) y = fmaxf(y, 0.f);
        C[(size_t)grow * KO + col] = y;
        if (Cbf) Cbf[(size_t)grow * KO + col] = (__bf16)y;
      }
    }
  }
}

// ---------------------------------------------------------------- GCN aggregation
__global__ void k_agg_init(const float* __restrict__ hw, const float* __restrict__ dinv,
                           const float* __restrict__ b, float* __restrict__ agg, int KO) {
  int i = blockIdx.x, c = threadIdx.x;
  float di = dinv[i];
  agg[(size_t)i * KO + c] = fmaf(hw[(size_t)i * KO + c], di * di, b[c]);
}

__global__ void k_conv_scatter(const long long* __restrict__ src,
                               const long long* __restrict__ dst,
                               const float* __restrict__ hw, const float* __restrict__ dinv,
                               float* __restrict__ agg, int KO) {
  int e = blockIdx.x, c = threadIdx.x;
  int s = (int)src[e], d = (int)dst[e];
  float w = dinv[s] * dinv[d];
  atomicAdd(&agg[(size_t)d * KO + c], hw[(size_t)s * KO + c] * w);
}

// ---------------------------------------------------------------- LN + residual + relu (HID=256)
// one wave32 per row; emits fp32 result + bf16 shadow (A-operand of next GEMM)
__global__ __launch_bounds__(256) void k_ln_res256(
    const float* __restrict__ agg, const float* __restrict__ ef,
    const long long* __restrict__ src,
    const float* __restrict__ g, const float* __restrict__ b,
    const float* __restrict__ res, float* __restrict__ out,
    __bf16* __restrict__ outb, int n) {
  int row = blockIdx.x * 8 + (threadIdx.x >> 5);
  int lane = threadIdx.x & 31;
  if (row >= n) return;
  int s = (int)src[row];
  float v[8];
  float mean = 0.f;
#pragma unroll
  for (int k = 0; k < 8; k++) {
    int c = lane + 32 * k;
    v[k] = agg[(size_t)row * 256 + c] + ef[(size_t)s * 256 + c];
    mean += v[k];
  }
#pragma unroll
  for (int o = 16; o; o >>= 1) mean += __shfl_xor(mean, o, 32);
  mean *= (1.f / 256.f);
  float var = 0.f;
#pragma unroll
  for (int k = 0; k < 8; k++) { float d0 = v[k] - mean; var += d0 * d0; }
#pragma unroll
  for (int o = 16; o; o >>= 1) var += __shfl_xor(var, o, 32);
  float rstd = rsqrtf(var * (1.f / 256.f) + LN_EPS);
#pragma unroll
  for (int k = 0; k < 8; k++) {
    int c = lane + 32 * k;
    float y = fmaf((v[k] - mean) * rstd, g[c], b[c]);
    y = fmaxf(y + res[(size_t)row * 256 + c], 0.f);
    out[(size_t)row * 256 + c] = y;
    outb[(size_t)row * 256 + c] = (__bf16)y;
  }
}

// ---------------------------------------------------------------- LN3 (OUT=128) + attn dot
__global__ __launch_bounds__(256) void k_ln3_out(
    const float* __restrict__ agg, const float* __restrict__ ef,
    const long long* __restrict__ src,
    const float* __restrict__ g, const float* __restrict__ b,
    const float* __restrict__ attn,
    float* __restrict__ out, float* __restrict__ dots, int n) {
  int row = blockIdx.x * 8 + (threadIdx.x >> 5);
  int lane = threadIdx.x & 31;
  if (row >= n) return;
  int s = (int)src[row];
  float v[4];
  float mean = 0.f;
#pragma unroll
  for (int k = 0; k < 4; k++) {
    int c = lane + 32 * k;
    v[k] = agg[(size_t)row * 128 + c] + ef[(size_t)s * 256 + c];  // ef_src[:, :128]
    mean += v[k];
  }
#pragma unroll
  for (int o = 16; o; o >>= 1) mean += __shfl_xor(mean, o, 32);
  mean *= (1.f / 128.f);
  float var = 0.f;
#pragma unroll
  for (int k = 0; k < 4; k++) { float d0 = v[k] - mean; var += d0 * d0; }
#pragma unroll
  for (int o = 16; o; o >>= 1) var += __shfl_xor(var, o, 32);
  float rstd = rsqrtf(var * (1.f / 128.f) + LN_EPS);
  float dot = 0.f;
#pragma unroll
  for (int k = 0; k < 4; k++) {
    int c = lane + 32 * k;
    float y = fmaf((v[k] - mean) * rstd, g[c], b[c]);
    out[(size_t)row * 128 + c] = y;
    dot = fmaf(y, attn[c], dot);
  }
#pragma unroll
  for (int o = 16; o; o >>= 1) dot += __shfl_xor(dot, o, 32);
  if (lane == 0) dots[row] = dot;
}

// ---------------------------------------------------------------- softmax over nodes
__global__ __launch_bounds__(256) void k_max_part(const float* __restrict__ d,
                                                  float* __restrict__ part, int n) {
  __shared__ float sm[256];
  float m = -3.4e38f;
  for (int i = blockIdx.x * 256 + threadIdx.x; i < n; i += 256 * gridDim.x)
    m = fmaxf(m, d[i]);
  sm[threadIdx.x] = m; __syncthreads();
  for (int o = 128; o; o >>= 1) {
    if (threadIdx.x < o) sm[threadIdx.x] = fmaxf(sm[threadIdx.x], sm[threadIdx.x + o]);
    __syncthreads();
  }
  if (threadIdx.x == 0) part[blockIdx.x] = sm[0];
}

__global__ __launch_bounds__(256) void k_max_final(const float* __restrict__ part,
                                                   float* __restrict__ scal) {
  __shared__ float sm[256];
  sm[threadIdx.x] = part[threadIdx.x]; __syncthreads();
  for (int o = 128; o; o >>= 1) {
    if (threadIdx.x < o) sm[threadIdx.x] = fmaxf(sm[threadIdx.x], sm[threadIdx.x + o]);
    __syncthreads();
  }
  if (threadIdx.x == 0) scal[0] = sm[0];
}

__global__ __launch_bounds__(256) void k_sum_part(const float* __restrict__ d,
                                                  const float* __restrict__ scal,
                                                  float* __restrict__ part, int n) {
  __shared__ float sm[256];
  float mx = scal[0];
  float acc = 0.f;
  for (int i = blockIdx.x * 256 + threadIdx.x; i < n; i += 256 * gridDim.x)
    acc += __expf(d[i] - mx);
  sm[threadIdx.x] = acc; __syncthreads();
  for (int o = 128; o; o >>= 1) {
    if (threadIdx.x < o) sm[threadIdx.x] += sm[threadIdx.x + o];
    __syncthreads();
  }
  if (threadIdx.x == 0) part[blockIdx.x] = sm[0];
}

__global__ __launch_bounds__(256) void k_sum_final(const float* __restrict__ part,
                                                   float* __restrict__ scal) {
  __shared__ float sm[256];
  sm[threadIdx.x] = part[threadIdx.x]; __syncthreads();
  for (int o = 128; o; o >>= 1) {
    if (threadIdx.x < o) sm[threadIdx.x] += sm[threadIdx.x + o];
    __syncthreads();
  }
  if (threadIdx.x == 0) scal[1] = sm[0];
}

__global__ __launch_bounds__(256) void k_wts(float* __restrict__ dots,
                                             const float* __restrict__ scal, int n) {
  int i = blockIdx.x * 256 + threadIdx.x;
  if (i < n) dots[i] = __expf(dots[i] - scal[0]) / scal[1];
}

__global__ __launch_bounds__(256) void k_scale(float* __restrict__ out,
                                               const float* __restrict__ dots, int n) {
  int i = blockIdx.x * 256 + threadIdx.x;
  if (i < n) out[i] *= dots[i >> 7];
}

// ---------------------------------------------------------------- launch
extern "C" void kernel_launch(void* const* d_in, const int* in_sizes, int n_in,
                              void* d_out, int out_size, void* d_ws, size_t ws_size,
                              hipStream_t stream) {
  (void)n_in; (void)out_size; (void)ws_size;
  const float*     x    = (const float*)d_in[0];
  const long long* ei   = (const long long*)d_in[1];
  const float*     ea   = (const float*)d_in[2];
  const float*     We1  = (const float*)d_in[3];
  const float*     be1  = (const float*)d_in[4];
  const float*     We2  = (const float*)d_in[5];
  const float*     be2  = (const float*)d_in[6];
  const float*     Wp   = (const float*)d_in[7];
  const float*     bp   = (const float*)d_in[8];
  const float*     Wc1  = (const float*)d_in[9];
  const float*     bc1  = (const float*)d_in[10];
  const float*     Wc2  = (const float*)d_in[11];
  const float*     bc2  = (const float*)d_in[12];
  const float*     Wc3  = (const float*)d_in[13];
  const float*     bc3  = (const float*)d_in[14];
  const float*     g1   = (const float*)d_in[15];
  const float*     b1   = (const float*)d_in[16];
  const float*     g2   = (const float*)d_in[17];
  const float*     b2   = (const float*)d_in[18];
  const float*     g3   = (const float*)d_in[19];
  const float*     b3   = (const float*)d_in[20];
  const float*     attn = (const float*)d_in[21];

  const int NN = in_sizes[0] / 768;      // 100000
  const long long* src = ei;
  const long long* dst = ei + NN;
  float* out = (float*)d_out;

  // workspace layout
  const size_t S = (size_t)NN * 256;
  float* B_h   = (float*)d_ws;           // h, later x2
  float* B_ef  = B_h + S;                // ef
  float* B_t   = B_ef + S;               // hw1/hw2/hw3
  float* B_agg = B_t + S;                // aggregation buffer
  float* B_x1  = B_agg + S;              // x1
  // bf16 shadows (A operands of GEMMs) — base is 64B aligned (5S*4 % 64 == 0)
  __bf16* T_bf  = (__bf16*)(B_x1 + S);   // t (edge-mlp hidden)
  __bf16* H_bf  = T_bf + S;              // h, later x2
  __bf16* X1_bf = H_bf + S;              // x1
  __bf16* Wp_b  = X1_bf + S;             // bf16 weights
  __bf16* We2_b = Wp_b + 768 * 256;
  __bf16* Wc1_b = We2_b + 256 * 256;
  __bf16* Wc2_b = Wc1_b + 256 * 256;
  __bf16* Wc3_b = Wc2_b + 256 * 256;
  float* deg   = (float*)(Wc3_b + 256 * 128);  // N (becomes dinv in place)
  float* dots  = deg + NN;               // N
  float* part  = dots + NN;              // 256
  float* scal  = part + 256;             // 2

  dim3 blk(256);
  auto cdiv = [](int a, int b) { return (a + b - 1) / b; };

  // 1) weights -> bf16
  k_f32_to_bf16<<<cdiv(768 * 256, 256), blk, 0, stream>>>(Wp,  Wp_b,  768 * 256);
  k_f32_to_bf16<<<cdiv(256 * 256, 256), blk, 0, stream>>>(We2, We2_b, 256 * 256);
  k_f32_to_bf16<<<cdiv(256 * 256, 256), blk, 0, stream>>>(Wc1, Wc1_b, 256 * 256);
  k_f32_to_bf16<<<cdiv(256 * 256, 256), blk, 0, stream>>>(Wc2, Wc2_b, 256 * 256);
  k_f32_to_bf16<<<cdiv(256 * 128, 256), blk, 0, stream>>>(Wc3, Wc3_b, 256 * 128);

  // 2) degrees (self-loop = 1), dinv = rsqrt(deg)
  k_fill1<<<cdiv(NN, 256), blk, 0, stream>>>(deg, NN);
  k_deg_scatter<<<cdiv(NN, 256), blk, 0, stream>>>(dst, deg, NN);
  k_rsqrt_<<<cdiv(NN, 256), blk, 0, stream>>>(deg, NN);
  float* dinv = deg;

  // 3) edge features: t = relu(ea@We1+be1) (bf16); ef = t@We2+be2 (fp32)
  k_edge_mlp1<<<cdiv(NN * 256, 256), blk, 0, stream>>>(ea, We1, be1, T_bf, NN);
  dim3 g256(cdiv(NN, 128), 4);
  dim3 g128(cdiv(NN, 128), 2);
  k_gemm<true><<<g256, blk, 0, stream>>>(T_bf, We2_b, be2, B_ef, nullptr, NN, 256, 256, 0);

  // 4) projection: h = x@Wp+bp (fp32 A path; emits bf16 shadow for conv1)
  k_gemm<false><<<g256, blk, 0, stream>>>(x, Wp_b, bp, B_h, H_bf, NN, 768, 256, 0);

  // 5) layer 1
  k_gemm<true><<<g256, blk, 0, stream>>>(H_bf, Wc1_b, nullptr, B_t, nullptr, NN, 256, 256, 0);
  k_agg_init<<<NN, 256, 0, stream>>>(B_t, dinv, bc1, B_agg, 256);
  k_conv_scatter<<<NN, 256, 0, stream>>>(src, dst, B_t, dinv, B_agg, 256);
  k_ln_res256<<<cdiv(NN, 8), blk, 0, stream>>>(B_agg, B_ef, src, g1, b1, B_h, B_x1, X1_bf, NN);

  // 6) layer 2 (x2 overwrites h's buffers)
  k_gemm<true><<<g256, blk, 0, stream>>>(X1_bf, Wc2_b, nullptr, B_t, nullptr, NN, 256, 256, 0);
  k_agg_init<<<NN, 256, 0, stream>>>(B_t, dinv, bc2, B_agg, 256);
  k_conv_scatter<<<NN, 256, 0, stream>>>(src, dst, B_t, dinv, B_agg, 256);
  k_ln_res256<<<cdiv(NN, 8), blk, 0, stream>>>(B_agg, B_ef, src, g2, b2, B_x1, B_h, H_bf, NN);

  // 7) layer 3 (OUT=128), x3 -> d_out, dots = x3@attn
  k_gemm<true><<<g128, blk, 0, stream>>>(H_bf, Wc3_b, nullptr, B_t, nullptr, NN, 256, 128, 0);
  k_agg_init<<<NN, 128, 0, stream>>>(B_t, dinv, bc3, B_agg, 128);
  k_conv_scatter<<<NN, 128, 0, stream>>>(src, dst, B_t, dinv, B_agg, 128);
  k_ln3_out<<<cdiv(NN, 8), blk, 0, stream>>>(B_agg, B_ef, src, g3, b3, attn, out, dots, NN);

  // 8) softmax over nodes + scale output
  k_max_part<<<256, blk, 0, stream>>>(dots, part, NN);
  k_max_final<<<1, blk, 0, stream>>>(part, scal);
  k_sum_part<<<256, blk, 0, stream>>>(dots, scal, part, NN);
  k_sum_final<<<1, blk, 0, stream>>>(part, scal);
  k_wts<<<cdiv(NN, 256), blk, 0, stream>>>(dots, scal, NN);
  k_scale<<<cdiv(NN * 128, 256), blk, 0, stream>>>(out, dots, NN * 128);
}
